// CausalAttention_57518202028183
// MI455X (gfx1250) — matrix operand order repaired and checked by reading, not compiled, not run on hardware
//
#include <hip/hip_runtime.h>

// ---------------------------------------------------------------------------
// Causal attention for MI455X (gfx1250, wave32, WMMA).
//   B=4, S=2048, D=768.  ~55 GFLOP vs ~60MB HBM traffic => compute bound.
//   All GEMMs use v_wmma_f32_16x16x32_bf16 (bf16 operands, f32 accumulate).
//   Intermediates (scores f32 67MB, P bf16 33MB, qkv bf16 38MB) live in the
//   192MB global L2 via the workspace.
// ---------------------------------------------------------------------------

#define BATCH 4
#define SEQ   2048
#define DIM   768

typedef __attribute__((ext_vector_type(16))) __bf16       v16bf;
typedef __attribute__((ext_vector_type(8)))  float        v8f;
typedef __attribute__((ext_vector_type(8)))  unsigned int v8u;
typedef __attribute__((ext_vector_type(4)))  unsigned int v4u;

#define DEV __device__ __forceinline__

// float -> bf16 round-to-nearest-even (finite inputs only)
DEV unsigned short f2bf(float f) {
  unsigned int u = __builtin_bit_cast(unsigned int, f);
  u += 0x7FFFu + ((u >> 16) & 1u);
  return (unsigned short)(u >> 16);
}

// A-fragment (16x32 bf16, M x K): lane l = row (l&15); K-chunks at
// base=(l>>4)*8 : elems [base..base+7] then [base+16..base+23].
DEV v16bf load_a16(const unsigned short* p) {
  v4u lo = *(const v4u*)p;        // 8 bf16
  v4u hi = *(const v4u*)(p + 16); // 8 bf16, K+16
  v8u t;
  t[0]=lo[0]; t[1]=lo[1]; t[2]=lo[2]; t[3]=lo[3];
  t[4]=hi[0]; t[5]=hi[1]; t[6]=hi[2]; t[7]=hi[3];
  return __builtin_bit_cast(v16bf, t);
}

// B-fragment (32x16 bf16, K x N): lane l = column (l&15); K rows (stride ld) at
// base=(l>>4)*8 : [base..base+7] then [base+16..base+23].
DEV v16bf load_b16(const unsigned short* p, long ld) {
  v16bf t;
#pragma unroll
  for (int j = 0; j < 8; ++j) {
    t[j]     = __builtin_bit_cast(__bf16, p[(long)j * ld]);
    t[8 + j] = __builtin_bit_cast(__bf16, p[(long)(16 + j) * ld]);
  }
  return t;
}

// pack 8 f32 -> 8 bf16 (memory order elem0..7)
DEV v4u pack8(v8f v) {
  v4u r;
#pragma unroll
  for (int j = 0; j < 4; ++j)
    r[j] = (unsigned int)f2bf(v[2*j]) | ((unsigned int)f2bf(v[2*j+1]) << 16);
  return r;
}

DEV v8f wmma_bf16(v16bf a, v16bf b, v8f c) {
  return __builtin_amdgcn_wmma_f32_16x16x32_bf16(false, a, false, b,
                                                 (short)0, c, false, false);
}

DEV v8f vzero8() { v8f z = {0.f,0.f,0.f,0.f,0.f,0.f,0.f,0.f}; return z; }

// ------------------------------- prep ---------------------------------------

__global__ void k_conv_x(const float* __restrict__ x,
                         unsigned short* __restrict__ xb, int n) {
  int i = blockIdx.x * 256 + threadIdx.x;
  if (i < n) xb[i] = f2bf(x[i]);
}

// wt[d][e] = w[e][d]  (weights are [d_out,d_in]; we need d-major for B frags)
__global__ void k_conv_wT(const float* __restrict__ w,
                          unsigned short* __restrict__ wt) {
  int i = blockIdx.x * 256 + threadIdx.x;
  if (i < DIM * DIM) {
    int d = i / DIM, e = i - d * DIM;
    wt[i] = f2bf(w[e * DIM + d]);
  }
}

// ------------------------------- QKV GEMM -----------------------------------
// Y[s][e] = sum_d X[s][d] * W[e][d] = sum_d xb[s][d] * wt[d][e]
// z=0 -> qb row-major, z=1 -> kt transposed [b][e][s], z=2 -> vb row-major.
// block = 128 (4 waves); each wave computes a 16(M) x 64(N) strip.
__global__ void k_qkv(const unsigned short* __restrict__ xb,
                      const unsigned short* __restrict__ wtq,
                      const unsigned short* __restrict__ wtk,
                      const unsigned short* __restrict__ wtv,
                      unsigned short* __restrict__ qb,
                      unsigned short* __restrict__ ktb,
                      unsigned short* __restrict__ vb) {
  const int z    = blockIdx.z;
  const int wave = threadIdx.x >> 5;
  const int lane = threadIdx.x & 31;
  const int r    = lane & 15;
  const int hi   = lane >> 4;
  const int ab   = hi * 8;
  const int s0   = blockIdx.y * 16;
  const int e0   = blockIdx.x * 256 + wave * 64;
  const unsigned short* wt = (z == 0) ? wtq : (z == 1) ? wtk : wtv;

  v8f acc[4];
#pragma unroll
  for (int nt = 0; nt < 4; ++nt) acc[nt] = vzero8();

  for (int d0 = 0; d0 < DIM; d0 += 32) {
    v16bf a = load_a16(xb + (long)(s0 + r) * DIM + d0 + ab);
    const unsigned short* wrow = wt + (long)(d0 + ab) * DIM + e0 + r;
#pragma unroll
    for (int nt = 0; nt < 4; ++nt)
      acc[nt] = wmma_bf16(a, load_b16(wrow + nt * 16, DIM), acc[nt]);
  }

  if (z == 1) { // K stored transposed: kt[b][e][s]; 8 rows/lane contiguous in s
    int b = s0 >> 11, sk0 = s0 & (SEQ - 1);
#pragma unroll
    for (int nt = 0; nt < 4; ++nt) {
      unsigned short* p = ktb + ((long)b * DIM + e0 + nt * 16 + r) * SEQ
                              + sk0 + 8 * hi;
      *(v4u*)p = pack8(acc[nt]);
    }
  } else {
    unsigned short* out = (z == 0) ? qb : vb;
#pragma unroll
    for (int nt = 0; nt < 4; ++nt)
#pragma unroll
      for (int j = 0; j < 8; ++j)
        out[(long)(s0 + j + 8 * hi) * DIM + e0 + nt * 16 + r] = f2bf(acc[nt][j]);
  }
}

// ------------------------------- scores QK^T --------------------------------
// scores[b][sq][sk] = (1/sqrt(D)) * sum_e q[sq][e] * k[sk][e], causal tiles
// only (kt <= qt). block=256 (8 waves); wave w handles kt = w, w+8, ...
// The 24 A-fragments of the Q tile are preloaded once (192 VGPRs) and reused.
__global__ void k_scores(const unsigned short* __restrict__ qb,
                         const unsigned short* __restrict__ ktb,
                         float* __restrict__ scores) {
  const int qt   = blockIdx.x;
  const int b    = blockIdx.y;
  const int wave = threadIdx.x >> 5;
  const int lane = threadIdx.x & 31;
  if (wave > qt) return;
  const int r  = lane & 15;
  const int hi = lane >> 4;
  const int ab = hi * 8;
  const int q0 = qt * 16;
  const float sc = 0.03608439182435161f; // 1/sqrt(768)

  v16bf A[24];
#pragma unroll
  for (int s = 0; s < 24; ++s)
    A[s] = load_a16(qb + (long)(b * SEQ + q0 + r) * DIM + s * 32 + ab);

  for (int kt = wave; kt <= qt; kt += 8) {
    v8f acc = vzero8();
#pragma unroll
    for (int s = 0; s < 24; ++s) {
      const unsigned short* kp =
          ktb + ((long)b * DIM + s * 32 + ab) * SEQ + kt * 16 + r;
      acc = wmma_bf16(A[s], load_b16(kp, SEQ), acc);
    }
#pragma unroll
    for (int j = 0; j < 8; ++j)
      scores[((long)b * SEQ + q0 + j + 8 * hi) * SEQ + kt * 16 + r] =
          acc[j] * sc;
  }
}

// ------------------------------- softmax ------------------------------------
// One wave per row. Causality by truncation: only cols [0..sq] are read.
// P is zero-padded up to a multiple of 32 so PV's K=32 steps are safe.
__global__ void k_softmax(const float* __restrict__ scores,
                          unsigned short* __restrict__ P) {
  const int row  = blockIdx.x * 8 + (threadIdx.x >> 5);
  const int lane = threadIdx.x & 31;
  const int sq   = row & (SEQ - 1);
  const int qt   = sq >> 4;
  const int nk32 = (((qt + 1) * 16) + 31) & ~31;
  const float* srow = scores + (long)row * SEQ;

  float m = -3.0e38f;
  for (int i = lane; i <= sq; i += 32) m = fmaxf(m, srow[i]);
#pragma unroll
  for (int off = 16; off; off >>= 1) m = fmaxf(m, __shfl_xor(m, off));

  float sum = 0.f;
  for (int i = lane; i <= sq; i += 32) sum += __expf(srow[i] - m);
#pragma unroll
  for (int off = 16; off; off >>= 1) sum += __shfl_xor(sum, off);
  const float inv = 1.0f / sum;

  unsigned short* prow = P + (long)row * SEQ;
  for (int i = lane; i < nk32; i += 32) {
    float p = (i <= sq) ? __expf(srow[i] - m) * inv : 0.f;
    prow[i] = f2bf(p);
  }
}

// ------------------------------- PV GEMM ------------------------------------
// O[b][sq][e] = sum_sk P[sq][sk] * V[sk][e]; only sk < nk32 (causal).
// block = 128 (4 waves); wave computes 16(M) x 64(N); output f32.
__global__ void k_pv(const unsigned short* __restrict__ P,
                     const unsigned short* __restrict__ vb,
                     float* __restrict__ out) {
  const int b    = blockIdx.z;
  const int qt   = blockIdx.y;
  const int wave = threadIdx.x >> 5;
  const int lane = threadIdx.x & 31;
  const int r    = lane & 15;
  const int hi   = lane >> 4;
  const int ab   = hi * 8;
  const int q0   = qt * 16;
  const int e0   = blockIdx.x * 256 + wave * 64;
  const int nsteps = (((qt + 1) * 16) + 31) / 32;

  v8f acc[4];
#pragma unroll
  for (int nt = 0; nt < 4; ++nt) acc[nt] = vzero8();

  for (int s = 0; s < nsteps; ++s) {
    const int sk0 = s * 32;
    v16bf a = load_a16(P + (long)(b * SEQ + q0 + r) * SEQ + sk0 + ab);
    const unsigned short* vrow = vb + (long)(b * SEQ + sk0 + ab) * DIM + e0 + r;
#pragma unroll
    for (int nt = 0; nt < 4; ++nt)
      acc[nt] = wmma_bf16(a, load_b16(vrow + nt * 16, DIM), acc[nt]);
  }
#pragma unroll
  for (int nt = 0; nt < 4; ++nt)
#pragma unroll
    for (int j = 0; j < 8; ++j)
      out[(long)(b * SEQ + q0 + j + 8 * hi) * DIM + e0 + nt * 16 + r] =
          acc[nt][j];
}

// ------------------------------- launcher -----------------------------------

extern "C" void kernel_launch(void* const* d_in, const int* in_sizes, int n_in,
                              void* d_out, int out_size, void* d_ws,
                              size_t ws_size, hipStream_t stream) {
  const float* x  = (const float*)d_in[0]; // [4,2048,768]
  const float* wq = (const float*)d_in[1]; // [768,768]
  const float* wk = (const float*)d_in[2];
  const float* wv = (const float*)d_in[3];
  float* out = (float*)d_out;

  char* ws = (char*)d_ws;
  const size_t SZ_XB  = (size_t)BATCH * SEQ * DIM * 2;   // 12.6 MB
  const size_t SZ_WT  = (size_t)DIM * DIM * 2;           // 1.2 MB each
  const size_t SZ_QKV = (size_t)BATCH * SEQ * DIM * 2;   // 12.6 MB each
  const size_t SZ_SC  = (size_t)BATCH * SEQ * SEQ * 4;   // 67 MB
  unsigned short* xb  = (unsigned short*)(ws);
  unsigned short* wtq = (unsigned short*)(ws + SZ_XB);
  unsigned short* wtk = (unsigned short*)(ws + SZ_XB + SZ_WT);
  unsigned short* wtv = (unsigned short*)(ws + SZ_XB + 2 * SZ_WT);
  unsigned short* qb  = (unsigned short*)(ws + SZ_XB + 3 * SZ_WT);
  unsigned short* ktb = (unsigned short*)(ws + SZ_XB + 3 * SZ_WT + SZ_QKV);
  unsigned short* vb  = (unsigned short*)(ws + SZ_XB + 3 * SZ_WT + 2 * SZ_QKV);
  float*          sc  = (float*)(ws + SZ_XB + 3 * SZ_WT + 3 * SZ_QKV);
  unsigned short* P   = (unsigned short*)(ws + SZ_XB + 3 * SZ_WT + 3 * SZ_QKV + SZ_SC);

  // 1) convert x to bf16; transpose+convert the three weight matrices
  {
    int n = BATCH * SEQ * DIM;
    k_conv_x<<<(n + 255) / 256, 256, 0, stream>>>(x, xb, n);
    int nw = (DIM * DIM + 255) / 256;
    k_conv_wT<<<nw, 256, 0, stream>>>(wq, wtq);
    k_conv_wT<<<nw, 256, 0, stream>>>(wk, wtk);
    k_conv_wT<<<nw, 256, 0, stream>>>(wv, wtv);
  }

  // 2) Q, K(transposed), V projections (WMMA bf16)
  {
    dim3 grid(DIM / 256, (BATCH * SEQ) / 16, 3); // (3, 512, 3)
    k_qkv<<<grid, 128, 0, stream>>>(xb, wtq, wtk, wtv, qb, ktb, vb);
  }

  // 3) causal score tiles (WMMA bf16, f32 out, scaled)
  {
    dim3 grid(SEQ / 16, BATCH); // (128, 4)
    k_scores<<<grid, 256, 0, stream>>>(qb, ktb, sc);
  }

  // 4) row softmax -> P (bf16)
  k_softmax<<<(BATCH * SEQ) / 8, 256, 0, stream>>>(sc, P);

  // 5) O = P @ V (WMMA bf16, f32 output)
  {
    dim3 grid(DIM / 256, SEQ / 16, BATCH); // (3, 128, 4)
    k_pv<<<grid, 128, 0, stream>>>(P, vb, out);
  }
}
